// MultiHeadAttentionBlock_21028159881837
// MI455X (gfx1250) — matrix operand order repaired and checked by reading, not compiled, not run on hardware
//
#include <hip/hip_runtime.h>
#include <hip/hip_bf16.h>

// ---------------- vector types ----------------
typedef __attribute__((ext_vector_type(4)))  float    v4f;
typedef __attribute__((ext_vector_type(8)))  float    v8f;
typedef __attribute__((ext_vector_type(8)))  _Float16 v8h;
typedef __attribute__((ext_vector_type(16))) _Float16 v16h;

#define SEQ_L   2048
#define DMODEL  1024
#define NHEAD   16
#define DK      64
#define NBATCH  2

// ---------------- WMMA wrapper ----------------
__device__ __forceinline__ v8f wmma_f16(v16h a, v16h b, v8f c) {
  // 8 args: (neg_a, A, neg_b, B, c_mod, C, reuse_a, reuse_b)
  return __builtin_amdgcn_wmma_f32_16x16x32_f16(
      false, a, false, b, (short)0, c, false, false);
}

// ---------------- small helpers ----------------
__device__ __forceinline__ v8h cvt8(const float* p) {
  v4f a = *(const v4f*)p;
  v4f b = *(const v4f*)(p + 4);
  v8f f = __builtin_shufflevector(a, b, 0, 1, 2, 3, 4, 5, 6, 7);
  return __builtin_convertvector(f, v8h);
}
__device__ __forceinline__ v16h combine16(v8h lo, v8h hi) {
  return __builtin_shufflevector(lo, hi, 0, 1, 2, 3, 4, 5, 6, 7,
                                 8, 9, 10, 11, 12, 13, 14, 15);
}

// A fragment (16x32 f16 MxK): halves 0..7 at k=(lane/16)*8, 8..15 at +16.
__device__ __forceinline__ v16h lds_frag_a(const _Float16* rowp, int lane) {
  const _Float16* p = rowp + ((lane >> 4) << 3);
  return combine16(*(const v8h*)p, *(const v8h*)(p + 16));
}
// B fragment (32x16 f16 KxN): halves 0..7 at k=(lane/16)*8, 8..15 at +16.
__device__ __forceinline__ v16h lds_frag_b(const _Float16* rowp, int lane) {
  const _Float16* p = rowp + ((lane >> 4) << 3);
  return combine16(*(const v8h*)p, *(const v8h*)(p + 16));
}
// A fragment straight from global f16 (used for Q, loaded once per wave).
__device__ __forceinline__ v16h gbl_frag_a(const _Float16* base, int ld,
                                           int row, int k0, int lane) {
  const _Float16* p = base + (size_t)row * ld + k0 + ((lane >> 4) << 3);
  return combine16(*(const v8h*)p, *(const v8h*)(p + 16));
}

// LDS staging stores (8 halves = 16B).
__device__ __forceinline__ void stage8(const float* g, _Float16* l) {
  *(v8h*)l = cvt8(g);
}
__device__ __forceinline__ void stage8(const _Float16* g, _Float16* l) {
  *(v8h*)l = *(const v8h*)g;
}

// ---------------- CDNA5 async global->LDS copy (ASYNCcnt) ------------------
__device__ __forceinline__ void async_copy16(unsigned lds_off,
                                             const void* gptr) {
  unsigned long long a = (unsigned long long)gptr;
  asm volatile("global_load_async_to_lds_b128 %0, %1, off"
               :: "v"(lds_off), "v"(a)
               : "memory");
}
__device__ __forceinline__ void wait_async0() {
  asm volatile("s_wait_asynccnt 0x0" ::: "memory");
}
__device__ __forceinline__ void wait_async2() {
  asm volatile("s_wait_asynccnt 0x2" ::: "memory");
}

// ---------------- cross-lane row reductions (wave32, 16-lane halves) -------
__device__ __forceinline__ float rmax16(float x) {
  x = fmaxf(x, __shfl_xor(x, 1));
  x = fmaxf(x, __shfl_xor(x, 2));
  x = fmaxf(x, __shfl_xor(x, 4));
  x = fmaxf(x, __shfl_xor(x, 8));
  return x;
}
__device__ __forceinline__ float rsum16(float x) {
  x += __shfl_xor(x, 1);
  x += __shfl_xor(x, 2);
  x += __shfl_xor(x, 4);
  x += __shfl_xor(x, 8);
  return x;
}

// ---------------- NT GEMM: C[m,n] = sum_k A[m,k] * B[n,k] ------------------
// Block 256 thr (8 waves 4x2), block tile 128x64, k-step 32, LDS-staged.
// MODE 0: f16 out, qh/kh layout [B,H,L,DK];  MODE 1: f16 out, vhT [B,H,DK,L];
// MODE 2: f32 out, row-major [M,N].
template <typename TA, int MODE>
__global__ __launch_bounds__(256) void gemm_nt(const TA* __restrict__ A,
                                               const float* __restrict__ B,
                                               void* __restrict__ Cout,
                                               int M, int N, int K) {
  __shared__ __align__(16) _Float16 Asl[128][40];   // pad 8 halves vs banks
  __shared__ __align__(16) _Float16 Bsl[64][40];

  const int t = threadIdx.x;
  const int lane = t & 31, wid = t >> 5;
  const int wm = wid >> 1, wn = wid & 1;
  const int bm = blockIdx.y * 128, bn = blockIdx.x * 64;
  const int ra = lane & 15;

  v8f acc[2][2] = {};
  for (int k0 = 0; k0 < K; k0 += 32) {
    __syncthreads();                       // prior tile fully consumed
    {
      const int r0 = t >> 2, c0 = (t & 3) << 3;   // 64 rows x 4 chunks
      stage8(A + (size_t)(bm + r0) * K + k0 + c0, &Asl[r0][c0]);
      stage8(A + (size_t)(bm + 64 + r0) * K + k0 + c0, &Asl[64 + r0][c0]);
      stage8(B + (size_t)(bn + r0) * K + k0 + c0, &Bsl[r0][c0]);
      if (k0 + 32 < K) {
        __builtin_prefetch(A + (size_t)(bm + r0) * K + k0 + 32 + c0, 0, 1);
        __builtin_prefetch(A + (size_t)(bm + 64 + r0) * K + k0 + 32 + c0, 0, 1);
        __builtin_prefetch(B + (size_t)(bn + r0) * K + k0 + 32 + c0, 0, 1);
      }
    }
    __syncthreads();
    v16h a0 = lds_frag_a(&Asl[wm * 32 + ra][0], lane);
    v16h a1 = lds_frag_a(&Asl[wm * 32 + 16 + ra][0], lane);
    v16h b0 = lds_frag_b(&Bsl[wn * 32 + ra][0], lane);
    v16h b1 = lds_frag_b(&Bsl[wn * 32 + 16 + ra][0], lane);
    acc[0][0] = wmma_f16(a0, b0, acc[0][0]);
    acc[0][1] = wmma_f16(a0, b1, acc[0][1]);
    acc[1][0] = wmma_f16(a1, b0, acc[1][0]);
    acc[1][1] = wmma_f16(a1, b1, acc[1][1]);
  }

  const int m0 = bm + wm * 32, n0 = bn + wn * 32;
#pragma unroll
  for (int i = 0; i < 2; i++) {
#pragma unroll
    for (int j = 0; j < 2; j++) {
#pragma unroll
      for (int r = 0; r < 8; r++) {
        const int m = m0 + i * 16 + ((lane >> 4) << 3) + r;
        const int n = n0 + j * 16 + (lane & 15);
        const float val = acc[i][j][r];
        if (MODE == 2) {
          ((float*)Cout)[(size_t)m * N + n] = val;
        } else {
          const int b = m >> 11, l = m & (SEQ_L - 1);
          const int h = n >> 6, d = n & (DK - 1);
          _Float16* o = (_Float16*)Cout;
          if (MODE == 0)
            o[(((size_t)b * NHEAD + h) * SEQ_L + l) * DK + d] = (_Float16)val;
          else
            o[(((size_t)b * NHEAD + h) * DK + d) * SEQ_L + l] = (_Float16)val;
        }
      }
    }
  }
}

// ---------------- flash attention ------------------------------------------
// One block = (b,h) x 128 query rows; 8 waves x 16-query tiles.
// K/V tiles (32 keys) staged into LDS with async b128 copies, double-buffered.
__global__ __launch_bounds__(256) void attn_kernel(
    const _Float16* __restrict__ qh, const _Float16* __restrict__ kh,
    const _Float16* __restrict__ vhT, const int* __restrict__ mask,
    _Float16* __restrict__ aout) {
  const int t = threadIdx.x;
  const int lane = t & 31, wid = t >> 5;
  const int bh = blockIdx.y;
  const int b = bh >> 4, h = bh & 15;
  const int qr = blockIdx.x * 128 + wid * 16;

  const _Float16* qb = qh + (size_t)bh * SEQ_L * DK;   // [L, DK]
  const _Float16* kb = kh + (size_t)bh * SEQ_L * DK;   // [L, DK]
  const _Float16* vb = vhT + (size_t)bh * DK * SEQ_L;  // [DK, L]
  const int* mrow = mask + (size_t)b * SEQ_L;

  __shared__ __align__(16) _Float16 Ksl[2][32][72];    // 32 keys x 64 d (+pad)
  __shared__ __align__(16) _Float16 Vsl[2][64][40];    // 64 d x 32 keys (+pad)
  __shared__ __align__(16) _Float16 plds[8][16][32];   // per-wave P tile

  // issue async staging for key-tile `kt` into buffer `ib`
  auto stage_kv = [&](int kt, int ib) {
    {
      const int row = t >> 3, c = (t & 7) << 3;        // K: 32 rows x 8 chunks
      async_copy16((unsigned)(size_t)&Ksl[ib][row][c],
                   kb + (size_t)(kt + row) * DK + c);
    }
    {
      const int row = t >> 2, c = (t & 3) << 3;        // V: 64 rows x 4 chunks
      async_copy16((unsigned)(size_t)&Vsl[ib][row][c],
                   vb + (size_t)row * SEQ_L + kt + c);
    }
  };

  const v16h q0 = gbl_frag_a(qb, DK, qr + (lane & 15), 0, lane);
  const v16h q1 = gbl_frag_a(qb, DK, qr + (lane & 15), 32, lane);

  v8f o[4] = {};
  float mr[8], sr[8];
#pragma unroll
  for (int r = 0; r < 8; r++) { mr[r] = -1e30f; sr[r] = 0.f; }
  const float scale = 0.125f;              // 1/sqrt(64)

  stage_kv(0, 0);                          // prologue: tile 0 -> buf 0
  int ib = 0;
  for (int kt = 0; kt < SEQ_L; kt += 32, ib ^= 1) {
    if (kt + 32 < SEQ_L) {                 // prefetch next tile, then drain
      stage_kv(kt + 32, ib ^ 1);           //   current tile's 2 async ops
      wait_async2();
    } else {
      wait_async0();
    }
    __syncthreads();                       // all waves' copies visible

    const _Float16(*kl)[72] = Ksl[ib];
    const _Float16(*vl)[40] = Vsl[ib];

    // S = Q @ K^T for 16 queries x 32 keys (fragments from LDS)
    v8f s0 = {}, s1 = {};
    s0 = wmma_f16(q0, lds_frag_b(&kl[lane & 15][0], lane), s0);
    s0 = wmma_f16(q1, lds_frag_b(&kl[lane & 15][32], lane), s0);
    s1 = wmma_f16(q0, lds_frag_b(&kl[16 + (lane & 15)][0], lane), s1);
    s1 = wmma_f16(q1, lds_frag_b(&kl[16 + (lane & 15)][32], lane), s1);

    const int mk0 = mrow[kt + (lane & 15)];
    const int mk1 = mrow[kt + 16 + (lane & 15)];
#pragma unroll
    for (int r = 0; r < 8; r++) {
      s0[r] = mk0 ? s0[r] * scale : -10000.f;
      s1[r] = mk1 ? s1[r] * scale : -10000.f;
    }

    // online softmax (row stats across the 16-lane C-fragment halves)
    float fac[8];
#pragma unroll
    for (int r = 0; r < 8; r++) {
      float x = rmax16(fmaxf(s0[r], s1[r]));
      float mn = fmaxf(mr[r], x);
      fac[r] = __expf(mr[r] - mn);
      mr[r] = mn;
    }
    float p0[8], p1[8];
#pragma unroll
    for (int r = 0; r < 8; r++) {
      p0[r] = __expf(s0[r] - mr[r]);
      p1[r] = __expf(s1[r] - mr[r]);
      const float rs = rsum16(p0[r] + p1[r]);
      sr[r] = sr[r] * fac[r] + rs;
    }
#pragma unroll
    for (int j = 0; j < 4; j++)
#pragma unroll
      for (int r = 0; r < 8; r++) o[j][r] *= fac[r];

    // P tile: C layout -> LDS -> A-fragment layout (per-wave, DS in-order)
    const int rowb = (lane >> 4) << 3, col = lane & 15;
#pragma unroll
    for (int r = 0; r < 8; r++) {
      plds[wid][rowb + r][col] = (_Float16)p0[r];
      plds[wid][rowb + r][16 + col] = (_Float16)p1[r];
    }
    const int arow = lane & 15, kc = (lane >> 4) << 3;
    const v8h plo = *(const v8h*)&plds[wid][arow][kc];
    const v8h phi = *(const v8h*)&plds[wid][arow][16 + kc];
    const v16h pf = combine16(plo, phi);

    // O += P @ V   (V tile rows = d, columns = keys)
#pragma unroll
    for (int j = 0; j < 4; j++) {
      v16h vf = lds_frag_b(&vl[j * 16 + (lane & 15)][0], lane);
      o[j] = wmma_f16(pf, vf, o[j]);
    }
    __syncthreads();                       // reads done before buffer reuse
  }

#pragma unroll
  for (int r = 0; r < 8; r++) sr[r] = 1.f / sr[r];
#pragma unroll
  for (int j = 0; j < 4; j++) {
#pragma unroll
    for (int r = 0; r < 8; r++) {
      const int m = qr + ((lane >> 4) << 3) + r;
      const int d = j * 16 + (lane & 15);
      // concat-head layout [B, L, H*DK] = row-major A for the final GEMM
      aout[(((size_t)b * SEQ_L + m) * NHEAD + h) * DK + d] =
          (_Float16)(o[j][r] * sr[r]);
    }
  }
}

// ---------------- host launcher ----------------
extern "C" void kernel_launch(void* const* d_in, const int* in_sizes, int n_in,
                              void* d_out, int out_size, void* d_ws,
                              size_t ws_size, hipStream_t stream) {
  (void)in_sizes; (void)n_in; (void)out_size; (void)ws_size;
  const float* q   = (const float*)d_in[0];
  const float* k   = (const float*)d_in[1];
  const float* v   = (const float*)d_in[2];
  const int*   msk = (const int*)d_in[3];
  const float* w_q = (const float*)d_in[4];
  const float* w_k = (const float*)d_in[5];
  const float* w_v = (const float*)d_in[6];
  const float* w_o = (const float*)d_in[7];
  float* out = (float*)d_out;

  const size_t NEL = (size_t)NBATCH * SEQ_L * DMODEL;   // 4096*1024
  _Float16* qh   = (_Float16*)d_ws;
  _Float16* kh   = qh + NEL;
  _Float16* vhT  = kh + NEL;
  _Float16* aout = vhT + NEL;

  const int M = NBATCH * SEQ_L;   // 4096
  const int N = DMODEL;           // 1024
  const int K = DMODEL;           // 1024
  dim3 ggrid(N / 64, M / 128);    // (16, 32)
  dim3 gblk(256);

  gemm_nt<float, 0><<<ggrid, gblk, 0, stream>>>(q, w_q, qh, M, N, K);
  gemm_nt<float, 0><<<ggrid, gblk, 0, stream>>>(k, w_k, kh, M, N, K);
  gemm_nt<float, 1><<<ggrid, gblk, 0, stream>>>(v, w_v, vhT, M, N, K);

  attn_kernel<<<dim3(SEQ_L / 128, NBATCH * NHEAD), gblk, 0, stream>>>(
      qh, kh, vhT, msk, aout);

  gemm_nt<_Float16, 2><<<ggrid, gblk, 0, stream>>>(aout, w_o, out, M, N, K);
}
